// MyMATURE_STE_73065983640229
// MI455X (gfx1250) — matrix operand order repaired and checked by reading, not compiled, not run on hardware
//
#include <hip/hip_runtime.h>
#include <hip/hip_bf16.h>
#include <cstdint>

typedef __attribute__((ext_vector_type(16))) _Float16 v16h;
typedef __attribute__((ext_vector_type(8)))  float    v8f;

#define GAMMA_C 0.4f
#define WMMA16(a, b, c) __builtin_amdgcn_wmma_f32_16x16x32_f16(false, (a), false, (b), (short)0, (c), false, false)

namespace {
constexpr int Bn = 4096, Pn = 12, Qn = 12, Nn = 325, Dn = 64, Kn = 15, Sn = 60;
constexpr int TEDIM = 31, Fn = Nn + TEDIM;   // 356
constexpr int XH  = Fn + Dn;                 // 420
constexpr int LDA = 512;                     // LDS A stride (halves)

// workspace layout (units: halves), all offsets multiples of 16 (32B aligned)
constexpr int SZ_WK1  = 448 * 64;
constexpr int OFF_WK  = 0;
constexpr int SZ_WG1  = 480 * 256;
constexpr int OFF_WG  = OFF_WK + 2 * SZ_WK1;
constexpr int SZ_WE1  = 64 * 64;
constexpr int OFF_WE  = OFF_WG + 2 * SZ_WG1;
constexpr int OFF_WA  = OFF_WE + 2 * SZ_WE1;
constexpr int OFF_WLB = OFF_WA + 2 * SZ_WE1;     // stacked [Wl;0;Wb;0] as 128x64
constexpr int OFF_W1  = OFF_WLB + 128 * 64;
constexpr int OFF_W2  = OFF_W1 + 128 * 64;
constexpr int W3NP    = 3904;                    // 3900 padded to mult of 16
constexpr int OFF_W3  = OFF_W2 + 64 * 64;
}

__device__ __forceinline__ float sigm(float x) { return 1.f / (1.f + __expf(-x)); }

// A fragment: 16x32 f16 tile, documented CDNA5 layout. A is LDS row-major [16][LDA].
// Per lane this reads halves {base..base+7, base+16..base+23} -> two ds_load_b128.
__device__ __forceinline__ v16h load_a_frag(const _Float16* A, int k0, int lane) {
  int m   = lane & 15;
  int khi = (lane >> 4) & 1;
  v16h out;
#pragma unroll
  for (int e = 0; e < 16; ++e) {
    int v  = e >> 1;
    int kk = ((v >= 4) ? 16 : 0) + khi * 8 + ((v & 3) << 1) + (e & 1);
    out[e] = A[m * LDA + k0 + kk];
  }
  return out;
}

// B fragment: pre-swizzled in workspace as [kc][nt][lane][16 halves] -> one 32B load
__device__ __forceinline__ v16h load_b_frag(const _Float16* base, int ntiles, int kc,
                                            int nt, int lane) {
  const v16h* p = (const v16h*)base;
  return p[(size_t)(kc * ntiles + nt) * 32 + lane];
}

struct SharedBufs {
  _Float16* A;                 // 16 x LDA halves
  _Float16* M;                 // 16 x 15 x 60 (f16)
  float *h, *c;                // 16 x 64
  float *kb, *wb, *eb, *ab, *gb; // key 16x64, attn 16x16, e/a 16x64, g 16x256
};

__device__ void marn_cell(int p, const float* xsrc, const int* TE, int t, int b0,
                          const _Float16* ws, const float* bk, const float* bg,
                          const float* be, const float* ba, SharedBufs sb, bool do_update) {
  const int tid  = threadIdx.x;
  const int lane = tid & 31;
  const int wave = tid >> 5;

  // Phase A: A = [x(325) | onehot_day(7) | onehot_tod(24) | h(64) | 0...]
  for (int idx = tid; idx < 16 * LDA; idx += 128) {
    int row = idx >> 9, col = idx & (LDA - 1);
    int gb = b0 + row;
    float v = 0.f;
    if (col < Nn)          v = xsrc[((size_t)gb * Pn + t) * Nn + col];
    else if (col < Nn + 7) v = (TE[((size_t)gb * (Pn + 1) + t) * 2 + 0] == (col - Nn)) ? 1.f : 0.f;
    else if (col < Fn)     v = (TE[((size_t)gb * (Pn + 1) + t) * 2 + 1] == (col - (Nn + 7))) ? 1.f : 0.f;
    else if (col < XH)     v = sb.h[row * 64 + (col - Fn)];
    sb.A[idx] = (_Float16)v;
  }
  __syncthreads();

  // Phase B: k = tanh(xh @ Wk + bk)  [16x448]@[448x64], one 16-col tile per wave,
  // two independent accumulator chains for XDL pipelining.
  {
    const _Float16* Bw = ws + OFF_WK + p * SZ_WK1;
    v8f acc0 = {0.f, 0.f, 0.f, 0.f, 0.f, 0.f, 0.f, 0.f};
    v8f acc1 = {0.f, 0.f, 0.f, 0.f, 0.f, 0.f, 0.f, 0.f};
    for (int kc = 0; kc < 14; kc += 2) {
      acc0 = WMMA16(load_a_frag(sb.A, kc * 32, lane),       load_b_frag(Bw, 4, kc,     wave, lane), acc0);
      acc1 = WMMA16(load_a_frag(sb.A, (kc + 1) * 32, lane), load_b_frag(Bw, 4, kc + 1, wave, lane), acc1);
    }
    int n = lane & 15, mb = (lane >= 16) ? 8 : 0;
    int col = wave * 16 + n;
    float bias = (col < Sn) ? bk[col] : 0.f;
#pragma unroll
    for (int r = 0; r < 8; ++r)
      sb.kb[(mb + r) * 64 + col] = tanhf(acc0[r] + acc1[r] + bias);
  }
  __syncthreads();

  // Phase C: slot attention softmax + read vector (one row per thread 0..15);
  // read vector written straight into A[:, 420:480] for the g-GEMM.
  if (tid < 16) {
    int row = tid;
    float sc[Kn];
    float mx = -1e30f;
    for (int k = 0; k < Kn; ++k) {
      float s = 0.f;
      const _Float16* Mp = sb.M + (row * Kn + k) * Sn;
      for (int ss = 0; ss < Sn; ++ss) s += (float)Mp[ss] * sb.kb[row * 64 + ss];
      sc[k] = s; mx = fmaxf(mx, s);
    }
    float den = 0.f;
    for (int k = 0; k < Kn; ++k) { sc[k] = __expf(sc[k] - mx); den += sc[k]; }
    float inv = 1.f / den;
    for (int k = 0; k < Kn; ++k) sb.wb[row * 16 + k] = sc[k] * inv;
    for (int ss = 0; ss < Sn; ++ss) {
      float r = 0.f;
      for (int k = 0; k < Kn; ++k) r += sb.wb[row * 16 + k] * (float)sb.M[(row * Kn + k) * Sn + ss];
      sb.A[row * LDA + XH + ss] = (_Float16)r;
    }
  }
  __syncthreads();

  // Phase E: g = [x,h,r] @ Wg + bg  [16x480]@[480x256]; kc-outer with one A
  // fragment shared by 4 independent accumulator chains (4 N-tiles per wave).
  {
    const _Float16* Bw = ws + OFF_WG + p * SZ_WG1;
    v8f a0 = {0.f, 0.f, 0.f, 0.f, 0.f, 0.f, 0.f, 0.f};
    v8f a1 = a0, a2 = a0, a3 = a0;
    for (int kc = 0; kc < 15; ++kc) {
      v16h af = load_a_frag(sb.A, kc * 32, lane);
      a0 = WMMA16(af, load_b_frag(Bw, 16, kc, wave * 4 + 0, lane), a0);
      a1 = WMMA16(af, load_b_frag(Bw, 16, kc, wave * 4 + 1, lane), a1);
      a2 = WMMA16(af, load_b_frag(Bw, 16, kc, wave * 4 + 2, lane), a2);
      a3 = WMMA16(af, load_b_frag(Bw, 16, kc, wave * 4 + 3, lane), a3);
    }
    int n = lane & 15, mb = (lane >= 16) ? 8 : 0;
    auto epi = [&](const v8f& acc, int j) {
      int col = (wave * 4 + j) * 16 + n;
      float bias = bg[col];
#pragma unroll
      for (int r = 0; r < 8; ++r) sb.gb[(mb + r) * 256 + col] = acc[r] + bias;
    };
    epi(a0, 0); epi(a1, 1); epi(a2, 2); epi(a3, 3);
  }
  __syncthreads();

  // Phase F: LSTM gates -> h_new, c_new; stage h_new f16 in A[:, :64]
  for (int idx = tid; idx < 16 * 64; idx += 128) {
    int row = idx >> 6, d = idx & 63;
    float ig = sb.gb[row * 256 + d];
    float fg = sb.gb[row * 256 + 64 + d];
    float og = sb.gb[row * 256 + 128 + d];
    float cg = sb.gb[row * 256 + 192 + d];
    float cn = sigm(fg) * sb.c[idx] + sigm(ig) * tanhf(cg);
    float hn = sigm(og) * tanhf(cn);
    sb.c[idx] = cn; sb.h[idx] = hn;
    sb.A[row * LDA + d] = (_Float16)hn;
  }
  __syncthreads();

  // Phase G: e = sigmoid(h@We+be), a = tanh(h@Wa+ba)  [16x64]@[64x64];
  // one A fragment per kc feeds both GEMMs (2 independent chains).
  {
    int n = lane & 15, mb = (lane >= 16) ? 8 : 0;
    int col = wave * 16 + n;
    const _Float16* BwE = ws + OFF_WE + p * SZ_WE1;
    const _Float16* BwA = ws + OFF_WA + p * SZ_WE1;
    v8f accE = {0.f, 0.f, 0.f, 0.f, 0.f, 0.f, 0.f, 0.f};
    v8f accA = accE;
    for (int kc = 0; kc < 2; ++kc) {
      v16h af = load_a_frag(sb.A, kc * 32, lane);
      accE = WMMA16(af, load_b_frag(BwE, 4, kc, wave, lane), accE);
      accA = WMMA16(af, load_b_frag(BwA, 4, kc, wave, lane), accA);
    }
    float biasE = (col < Sn) ? be[col] : 0.f;
    float biasA = (col < Sn) ? ba[col] : 0.f;
#pragma unroll
    for (int r = 0; r < 8; ++r) {
      sb.eb[(mb + r) * 64 + col] = sigm(accE[r] + biasE);
      sb.ab[(mb + r) * 64 + col] = tanhf(accA[r] + biasA);
    }
  }
  __syncthreads();

  // Phase H (subject cell): in-place M update
  if (do_update) {
    for (int idx = tid; idx < 16 * Kn * Sn; idx += 128) {
      int row = idx / (Kn * Sn), rem = idx % (Kn * Sn);
      int k = rem / Sn, ss = rem % Sn;
      float mo = (float)sb.M[idx];
      float wv = sb.wb[row * 16 + k];
      sb.M[idx] = (_Float16)(mo + wv * (sb.ab[row * 64 + ss] - sb.eb[row * 64 + ss] * mo));
    }
    __syncthreads();
  }
}

__global__ __launch_bounds__(128)
void marn_main(const float* X, const float* Z, const int* TE,
               const float* bk, const float* bg, const float* be, const float* ba,
               const float* bK, const float* b1, const float* b2, const float* b3,
               const _Float16* ws, float* out) {
  __shared__ _Float16 A[16 * LDA];           // 16 KB
  __shared__ _Float16 Mr[16 * Kn * Sn];      // 28.8 KB
  __shared__ _Float16 Ms[16 * Kn * Sn];      // 28.8 KB
  __shared__ float gbuf[16 * 256];           // 16 KB
  __shared__ float hr[16 * 64], hs[16 * 64], cr[16 * 64], cs[16 * 64];
  __shared__ float kbuf[16 * 64];
  __shared__ float wbuf[16 * 16];
  __shared__ float ebuf[16 * 64], abuf[16 * 64];
  __shared__ float lbuf[16 * 64], bbuf[16 * 64];
  __shared__ float gatebuf[16 * 64];
  // total ~132 KB -> 2 workgroups resident per WGP (320 KB LDS)

  const int tid  = threadIdx.x;
  const int lane = tid & 31;
  const int wave = tid >> 5;
  const int b0   = blockIdx.x * 16;

  for (int idx = tid; idx < 16 * Kn * Sn; idx += 128) {
    Mr[idx] = (_Float16)0.f; Ms[idx] = (_Float16)0.f;
  }
  for (int idx = tid; idx < 16 * 64; idx += 128) {
    hr[idx] = 0.f; hs[idx] = 0.f; cr[idx] = 0.f; cs[idx] = 0.f; lbuf[idx] = 0.f; bbuf[idx] = 0.f;
  }
  __syncthreads();

  SharedBufs sbr{A, Mr, hr, cr, kbuf, wbuf, ebuf, abuf, gbuf};
  SharedBufs sbs{A, Ms, hs, cs, kbuf, wbuf, ebuf, abuf, gbuf};

  for (int t = 0; t < Pn; ++t) {
    // reference cell (input Z); defer its memory update (need M_r_old for delta)
    marn_cell(0, Z, TE, t, b0, ws, bk, bg, be, ba, sbr, false);

    // KAM: delta stats -> l2, bb2; stage [l2 | bb2] into A[:, :128]
    for (int idx = tid; idx < 16 * Sn; idx += 128) {
      int row = idx / Sn, ss = idx % Sn;
      float e = ebuf[row * 64 + ss], a = abuf[row * 64 + ss];
      float sa = 0.f, sd = 0.f;
      for (int k = 0; k < Kn; ++k) {
        float mo = (float)Mr[(row * Kn + k) * Sn + ss];
        float d  = wbuf[row * 16 + k] * (a - e * mo);
        sa += fabsf(d); sd += d;
      }
      float l2 = GAMMA_C * lbuf[row * 64 + ss] + (1.f - GAMMA_C) * sa * (1.f / Kn);
      float v2 = GAMMA_C * bbuf[row * 64 + ss] + (1.f - GAMMA_C) * sd * (1.f / Kn);
      lbuf[row * 64 + ss] = l2; bbuf[row * 64 + ss] = v2;
      A[row * LDA + ss]      = (_Float16)l2;
      A[row * LDA + 64 + ss] = (_Float16)v2;
    }
    for (int idx = tid; idx < 16 * 4; idx += 128) {
      int row = idx >> 2, j = idx & 3;
      A[row * LDA + Sn + j]      = (_Float16)0.f;
      A[row * LDA + 64 + Sn + j] = (_Float16)0.f;
    }
    __syncthreads();

    // gate = sigmoid([l2|bb2] @ [Wl;Wb] + bK)   [16x128]@[128x64]
    {
      const _Float16* Bw = ws + OFF_WLB;
      v8f acc0 = {0.f, 0.f, 0.f, 0.f, 0.f, 0.f, 0.f, 0.f};
      v8f acc1 = acc0;
      for (int kc = 0; kc < 4; kc += 2) {
        acc0 = WMMA16(load_a_frag(A, kc * 32, lane),       load_b_frag(Bw, 4, kc,     wave, lane), acc0);
        acc1 = WMMA16(load_a_frag(A, (kc + 1) * 32, lane), load_b_frag(Bw, 4, kc + 1, wave, lane), acc1);
      }
      int n = lane & 15, mb = (lane >= 16) ? 8 : 0;
      int col = wave * 16 + n;
      float bias = (col < Sn) ? bK[col] : 0.f;
#pragma unroll
      for (int r = 0; r < 8; ++r)
        gatebuf[(mb + r) * 64 + col] = sigm(acc0[r] + acc1[r] + bias);
    }
    __syncthreads();

    // M_s += gate * delta ; M_r += delta (delta recomputed from M_r_old)
    for (int idx = tid; idx < 16 * Kn * Sn; idx += 128) {
      int row = idx / (Kn * Sn), rem = idx % (Kn * Sn);
      int k = rem / Sn, ss = rem % Sn;
      float mo = (float)Mr[idx];
      float d  = wbuf[row * 16 + k] * (abuf[row * 64 + ss] - ebuf[row * 64 + ss] * mo);
      Ms[idx] = (_Float16)((float)Ms[idx] + gatebuf[row * 64 + ss] * d);
      Mr[idx] = (_Float16)(mo + d);
    }
    __syncthreads();

    // subject cell (input X) with in-place M update
    marn_cell(1, X, TE, t, b0, ws, bk + Sn, bg + 256, be + Sn, ba + Sn, sbs, true);
  }

  // Final MLP: relu([h_r|h_s]@W1+b1) -> relu(@W2+b2) -> @W3+b3
  for (int idx = tid; idx < 16 * 128; idx += 128) {
    int row = idx >> 7, col = idx & 127;
    float v = (col < 64) ? hr[row * 64 + col] : hs[row * 64 + (col - 64)];
    A[row * LDA + col] = (_Float16)v;
  }
  __syncthreads();
  {
    const _Float16* Bw = ws + OFF_W1;
    v8f acc0 = {0.f, 0.f, 0.f, 0.f, 0.f, 0.f, 0.f, 0.f};
    v8f acc1 = acc0;
    for (int kc = 0; kc < 4; kc += 2) {
      acc0 = WMMA16(load_a_frag(A, kc * 32, lane),       load_b_frag(Bw, 4, kc,     wave, lane), acc0);
      acc1 = WMMA16(load_a_frag(A, (kc + 1) * 32, lane), load_b_frag(Bw, 4, kc + 1, wave, lane), acc1);
    }
    int n = lane & 15, mb = (lane >= 16) ? 8 : 0;
    int col = wave * 16 + n;
    float bias = b1[col];
#pragma unroll
    for (int r = 0; r < 8; ++r)
      kbuf[(mb + r) * 64 + col] = fmaxf(acc0[r] + acc1[r] + bias, 0.f);
  }
  __syncthreads();
  for (int idx = tid; idx < 16 * 64; idx += 128) {
    int row = idx >> 6, col = idx & 63;
    A[row * LDA + col] = (_Float16)kbuf[idx];
  }
  __syncthreads();
  {
    const _Float16* Bw = ws + OFF_W2;
    v8f acc = {0.f, 0.f, 0.f, 0.f, 0.f, 0.f, 0.f, 0.f};
    for (int kc = 0; kc < 2; ++kc) {
      acc = WMMA16(load_a_frag(A, kc * 32, lane), load_b_frag(Bw, 4, kc, wave, lane), acc);
    }
    int n = lane & 15, mb = (lane >= 16) ? 8 : 0;
    int col = wave * 16 + n;
    float bias = b2[col];
#pragma unroll
    for (int r = 0; r < 8; ++r)
      gbuf[(mb + r) * 64 + col] = fmaxf(acc[r] + bias, 0.f);  // h2 staged in gbuf
  }
  __syncthreads();
  for (int idx = tid; idx < 16 * 64; idx += 128) {
    int row = idx >> 6, col = idx & 63;
    A[row * LDA + col] = (_Float16)gbuf[idx];
  }
  __syncthreads();
  {
    const _Float16* Bw = ws + OFF_W3;
    // A fragments are loop-invariant across the 61 output tiles of this wave.
    v16h af0 = load_a_frag(A, 0, lane);
    v16h af1 = load_a_frag(A, 32, lane);
    for (int nt = wave; nt < W3NP / 16; nt += 4) {
      v8f acc = {0.f, 0.f, 0.f, 0.f, 0.f, 0.f, 0.f, 0.f};
      acc = WMMA16(af0, load_b_frag(Bw, W3NP / 16, 0, nt, lane), acc);
      acc = WMMA16(af1, load_b_frag(Bw, W3NP / 16, 1, nt, lane), acc);
      int n = lane & 15, mb = (lane >= 16) ? 8 : 0;
      int col = nt * 16 + n;
      if (col < Qn * Nn) {
        float bias = b3[col];
#pragma unroll
        for (int r = 0; r < 8; ++r)
          out[(size_t)(b0 + mb + r) * (Qn * Nn) + col] = acc[r] + bias;
      }
    }
  }
}

// Convert f32 [Ks x Ns] weight -> padded [Kp x Np] f16 pre-swizzled into WMMA
// B-fragment order: [kc][nt][lane][16 halves]; lane = column (N), halves = K
// (lanes 0-15: K=0..7,16..23, lanes 16-31: K=8..15,24..31 within the 32-wide K-chunk).
__global__ void frag_pack(const float* src, _Float16* dst, int Ks, int Ns, int Kp, int Np) {
  int idx = blockIdx.x * blockDim.x + threadIdx.x;
  int tot = Kp * Np;
  if (idx >= tot) return;
  int e  = idx & 15;
  int l  = (idx >> 4) & 31;
  int fi = idx >> 9;
  int ntiles = Np >> 4;
  int nt = fi % ntiles;
  int kc = fi / ntiles;
  int khi = (l >> 4) & 1;
  int row = kc * 32 + ((e >> 3) << 4) + khi * 8 + (e & 7);
  int col = nt * 16 + (l & 15);
  float v = (row < Ks && col < Ns) ? src[(size_t)row * Ns + col] : 0.f;
  dst[idx] = (_Float16)v;
}

extern "C" void kernel_launch(void* const* d_in, const int* in_sizes, int n_in,
                              void* d_out, int out_size, void* d_ws, size_t ws_size,
                              hipStream_t stream) {
  (void)in_sizes; (void)n_in; (void)out_size; (void)ws_size;
  const float* X  = (const float*)d_in[0];
  const float* Z  = (const float*)d_in[1];
  const int*   TE = (const int*)d_in[2];
  const float* Wk = (const float*)d_in[3];
  const float* bk = (const float*)d_in[4];
  const float* Wg = (const float*)d_in[5];
  const float* bg = (const float*)d_in[6];
  const float* We = (const float*)d_in[7];
  const float* be = (const float*)d_in[8];
  const float* Wa = (const float*)d_in[9];
  const float* ba = (const float*)d_in[10];
  const float* Wl = (const float*)d_in[11];
  const float* Wb = (const float*)d_in[12];
  const float* bK = (const float*)d_in[13];
  const float* W1 = (const float*)d_in[14];
  const float* b1 = (const float*)d_in[15];
  const float* W2 = (const float*)d_in[16];
  const float* b2 = (const float*)d_in[17];
  const float* W3 = (const float*)d_in[18];
  const float* b3 = (const float*)d_in[19];
  float*    out = (float*)d_out;
  _Float16* ws  = (_Float16*)d_ws;

  auto pack = [&](const float* src, int dstOff, int Ks, int Ns, int Kp, int Np) {
    int tot = Kp * Np;
    frag_pack<<<(tot + 255) / 256, 256, 0, stream>>>(src, ws + dstOff, Ks, Ns, Kp, Np);
  };
  // MARN cell weights, both cells
  pack(Wk,            OFF_WK,           420, 60, 448, 64);
  pack(Wk + 420 * 60, OFF_WK + SZ_WK1,  420, 60, 448, 64);
  pack(Wg,             OFF_WG,          480, 256, 480, 256);
  pack(Wg + 480 * 256, OFF_WG + SZ_WG1, 480, 256, 480, 256);
  pack(We,           OFF_WE,           64, 60, 64, 64);
  pack(We + 64 * 60, OFF_WE + SZ_WE1,  64, 60, 64, 64);
  pack(Wa,           OFF_WA,           64, 60, 64, 64);
  pack(Wa + 64 * 60, OFF_WA + SZ_WE1,  64, 60, 64, 64);
  // KAM gate: stacked [Wl (rows 0-59); zeros; Wb (rows 64-123); zeros] as 128x64
  pack(Wl, OFF_WLB,           60, 60, 64, 64);
  pack(Wb, OFF_WLB + 64 * 64, 60, 60, 64, 64);
  // MLP head
  pack(W1, OFF_W1, 128, 64, 128, 64);
  pack(W2, OFF_W2, 64, 64, 64, 64);
  pack(W3, OFF_W3, 64, 3900, 64, W3NP);

  marn_main<<<Bn / 16, 128, 0, stream>>>(X, Z, TE, bk, bg, be, ba, bK, b1, b2, b3, ws, out);
}
